// GNN_NCM_14542759264927
// MI455X (gfx1250) — matrix operand order repaired and checked by reading, not compiled, not run on hardware
//
#include <hip/hip_runtime.h>

typedef __attribute__((ext_vector_type(16))) _Float16 v16h;
typedef __attribute__((ext_vector_type(8)))  _Float16 v8h;
typedef __attribute__((ext_vector_type(8)))  float    v8f;

#define N_NODES 10000
#define N_EDGES 10000
#define F_IN    32
#define NOISE_D 4
#define DIN1    36   // F_IN + NOISE
#define EDGE_WAVES 4

static_assert(N_EDGES % EDGE_WAVES == 0, "one edge per wave");

// ---------------------------------------------------------------------------
// gfx1250 async-copy helpers (ASYNCcnt path; per-lane 16B global->LDS DMA).
// ---------------------------------------------------------------------------
__device__ __forceinline__ void gnn_async_b128(uint32_t lds_byte, uint64_t gaddr) {
  asm volatile("global_load_async_to_lds_b128 %0, %1, off"
               :: "v"(lds_byte), "v"(gaddr) : "memory");
}
__device__ __forceinline__ void gnn_wait_async0() {
  asm volatile("s_wait_asynccnt 0x0" ::: "memory");
}
__device__ __forceinline__ void gnn_wait_ds0() {
  asm volatile("s_wait_dscnt 0x0" ::: "memory");
}

// ---------------------------------------------------------------------------
// Build h0 (f16, [N,64]): cols 0..31 = x, 32..35 = noise, 36..63 = 0 pad.
// ---------------------------------------------------------------------------
__global__ void gnn_prep_h0(const float* __restrict__ x,
                            const float* __restrict__ noise,
                            _Float16* __restrict__ h0) {
  int i = blockIdx.x * 256 + threadIdx.x;
  if (i >= N_NODES * 64) return;
  int n = i >> 6, c = i & 63;
  float v = (c < F_IN) ? x[n * F_IN + c]
          : (c < DIN1) ? noise[n * NOISE_D + (c - F_IN)]
          : 0.f;
  h0[i] = (_Float16)v;
}

// ---------------------------------------------------------------------------
// Pre-permute shared update weight Uw [Kreal,64] f32 into the GEMM's contiguous-k WMMA
// B-fragment order: uwp[((ks*4+nt)*32+lane)*16 + m] = Uw[ks*32+kb+m, nt*16+n]
// so the GEMM loads each B tile as one aligned 32B vector.
// ---------------------------------------------------------------------------
__global__ void gnn_prep_uwfrag(const float* __restrict__ Uw, int Kreal,
                                _Float16* __restrict__ uwp) {
  int i = blockIdx.x * 256 + threadIdx.x;     // 4*4*32*16 = 8192 total
  if (i >= 8192) return;
  int m    = i & 15;
  int lane = (i >> 4) & 31;
  int nt   = (i >> 9) & 3;
  int ks   = (i >> 11) & 3;
  int k    = ks * 32 + ((lane >> 4) << 4) + m;
  int col  = nt * 16 + (lane & 15);
  uwp[i] = (_Float16)((k < Kreal) ? Uw[k * 64 + col] : 0.f);
}

// ---------------------------------------------------------------------------
// Build the single-row A fragments (x in matrix row 0; lanes 0/16 carry data).
// ---------------------------------------------------------------------------
__device__ __forceinline__ void gnn_build_rowA(const _Float16* __restrict__ xr,
                                               int lane, v16h& A0, v16h& A1) {
  A0 = {}; A1 = {};
  if ((lane & 15) == 0) {
    const int klo = (lane >> 4) * 16;         // lane 0 -> 0, lane 16 -> 16
    v8h p0 = *(const v8h*)(xr + klo);
    v8h p1 = *(const v8h*)(xr + klo + 8);
    v8h p2 = *(const v8h*)(xr + klo + 32);
    v8h p3 = *(const v8h*)(xr + klo + 40);
    A0 = __builtin_shufflevector(p0, p1, 0,1,2,3,4,5,6,7,8,9,10,11,12,13,14,15);
    A1 = __builtin_shufflevector(p2, p3, 0,1,2,3,4,5,6,7,8,9,10,11,12,13,14,15);
  }
}

// ---------------------------------------------------------------------------
// Row-vector [1xKR] @ W[KRx64] on v_wmma_f32_16x16x32_f16, streaming W through
// a per-wave LDS stage with async b128 DMA (one 32x64 f32 K-slab at a time).
// acc[nt][0] on lanes 0..15 receives output element nt*16+lane.
// ---------------------------------------------------------------------------
template <int KR>
__device__ __forceinline__ void gnn_gemv_staged(const float* __restrict__ W,
                                                float* __restrict__ stage,
                                                uint32_t stage_lds,
                                                const v16h& A0, const v16h& A1,
                                                int lane, v8f acc[4]) {
  const int n  = lane & 15;
  const int kb = (lane >> 4) << 4;            // 0 / 16 per half-wave
#pragma unroll
  for (int s = 0; s < 2; ++s) {
    const int rows  = (KR - s * 32) < 32 ? (KR - s * 32) : 32;
    const int bytes = rows * 256;             // 64 f32 per row
    const uint64_t g = (uint64_t)(uintptr_t)(W + s * 32 * 64);
    gnn_wait_ds0();                           // stage reuse: prior LDS reads done
#pragma unroll
    for (int it = 0; it < bytes / 512; ++it) {// 512B per wave-wide b128 copy
      const uint32_t t = (uint32_t)(it * 512 + lane * 16);
      gnn_async_b128(stage_lds + t, g + t);
    }
    gnn_wait_async0();
#pragma unroll
    for (int nt = 0; nt < 4; ++nt) {
      v16h b;
#pragma unroll
      for (int j = 0; j < 8; ++j) {
        const int k0 = s * 32 + kb + 2 * j;
        b[2*j]   = (_Float16)((k0     < KR) ? stage[(kb + 2*j    ) * 64 + nt*16 + n] : 0.f);
        b[2*j+1] = (_Float16)((k0 + 1 < KR) ? stage[(kb + 2*j + 1) * 64 + nt*16 + n] : 0.f);
      }
      acc[nt] = __builtin_amdgcn_wmma_f32_16x16x32_f16(false, s ? A1 : A0, false, b,
                                                       (short)0, acc[nt], false, false);
    }
  }
}

// ---------------------------------------------------------------------------
// Per-edge MLP: m = relu(x_src @ W1 + b1); out = m @ W2 + b2; aggr[dst] += out
// One edge per wave, 4 waves / block; all W traffic via async DMA.
// ---------------------------------------------------------------------------
template <int KR1>
__global__ __launch_bounds__(EDGE_WAVES * 32) void gnn_edge_layer(
    const _Float16* __restrict__ hin,   // [N,64] f16 (zero padded past KR1)
    const int* __restrict__ ei,         // [2,E]
    const float* __restrict__ W1,       // [E,KR1,64]
    const float* __restrict__ b1,       // [E,64]
    const float* __restrict__ W2,       // [E,64,64]
    const float* __restrict__ b2,       // [E,64]
    float* __restrict__ aggr)           // [N,64], pre-zeroed
{
  __shared__ __align__(16) float     stage[EDGE_WAVES][2048];  // 8KB per wave
  __shared__ __align__(16) _Float16  lm[EDGE_WAVES][64];
  const int lane = threadIdx.x & 31;
  const int wid  = threadIdx.x >> 5;
  const int e    = blockIdx.x * EDGE_WAVES + wid;
  const int src  = ei[e];
  const int dst  = ei[N_EDGES + e];
  const float* W1e = W1 + (size_t)e * KR1 * 64;
  const float* W2e = W2 + (size_t)e * 64 * 64;
  __builtin_prefetch(W2e, 0, 1);                       // warm L2 for 2nd GEMV
  float*   stg     = stage[wid];
  uint32_t stg_lds = (uint32_t)(uintptr_t)stg;         // low 32b = LDS offset

  // --- GEMV 1: x @ W1 ---
  v16h a0, a1;
  gnn_build_rowA(hin + (size_t)src * 64, lane, a0, a1);
  v8f acc[4] = {};
  gnn_gemv_staged<KR1>(W1e, stg, stg_lds, a0, a1, lane, acc);

  const int n = lane & 15;
  if (lane < 16) {
#pragma unroll
    for (int nt = 0; nt < 4; ++nt) {
      float m = acc[nt][0] + b1[(size_t)e * 64 + nt * 16 + n];
      lm[wid][nt * 16 + n] = (_Float16)fmaxf(m, 0.f);  // relu
    }
  }
  // same-wave DS ordering: store->load to lm is in order, no barrier needed

  // --- GEMV 2: m @ W2 ---
  v16h c0, c1;
  gnn_build_rowA(lm[wid], lane, c0, c1);
  v8f acc2[4] = {};
  gnn_gemv_staged<64>(W2e, stg, stg_lds, c0, c1, lane, acc2);

  if (lane < 16) {
#pragma unroll
    for (int nt = 0; nt < 4; ++nt) {
      float v = acc2[nt][0] + b2[(size_t)e * 64 + nt * 16 + n];
      atomicAdd(&aggr[(size_t)dst * 64 + nt * 16 + n], v);   // segment_sum
    }
  }
}

// ---------------------------------------------------------------------------
// Concat builders (f16, K padded to 128).
// ---------------------------------------------------------------------------
__global__ void gnn_cat1(const float* __restrict__ x, const float* __restrict__ noise,
                         const float* __restrict__ aggr, _Float16* __restrict__ hcat) {
  int i = blockIdx.x * 256 + threadIdx.x;
  if (i >= N_NODES * 128) return;
  int nd = i >> 7, c = i & 127;
  float v = (c < F_IN)      ? x[nd * F_IN + c]
          : (c < DIN1)      ? noise[nd * NOISE_D + (c - F_IN)]
          : (c < DIN1 + 64) ? aggr[(size_t)nd * 64 + (c - DIN1)]
          : 0.f;
  hcat[i] = (_Float16)v;
}

__global__ void gnn_cat2(const float* __restrict__ aggr, _Float16* __restrict__ hcat) {
  int i = blockIdx.x * 256 + threadIdx.x;
  if (i >= N_NODES * 64) return;
  int nd = i >> 6, c = i & 63;
  hcat[(size_t)nd * 128 + 64 + c] = (_Float16)aggr[i];
}

// ---------------------------------------------------------------------------
// Update MLP: relu(hcat[N,128] @ Uw[128,64] + Ub) -- true WMMA GEMM.
// Wave = 16 nodes x 64 outs; B tiles are single 32B loads from uwfrag.
// ---------------------------------------------------------------------------
__global__ __launch_bounds__(256) void gnn_update_layer(
    const _Float16* __restrict__ hcat,    // [N,128]
    const _Float16* __restrict__ uwfrag,  // [4][4][32][16] B fragments
    const float* __restrict__ Ub,         // [64]
    float* __restrict__ hout_f32,         // [N,64] or null
    _Float16* __restrict__ hout_f16,      // [N,64] or null
    _Float16* __restrict__ hcat_next)     // [N,128] first half, or null
{
  const int lane = threadIdx.x & 31;
  const int wid  = threadIdx.x >> 5;
  const int m0   = (blockIdx.x * 8 + wid) * 16;
  const int n    = lane & 15;
  const int row  = m0 + n;
  const bool rowok = row < N_NODES;
  const _Float16* hr = hcat + (size_t)row * 128;

  v8f acc[4] = {};
#pragma unroll
  for (int ks = 0; ks < 4; ++ks) {
    v16h a = {};
    if (rowok) {
      const int klo = ks * 32 + ((lane >> 4) * 16);
      v8h lo = *(const v8h*)(hr + klo);
      v8h hi = *(const v8h*)(hr + klo + 8);
      a = __builtin_shufflevector(lo, hi, 0,1,2,3,4,5,6,7,8,9,10,11,12,13,14,15);
    }
#pragma unroll
    for (int nt = 0; nt < 4; ++nt) {
      v16h b = *(const v16h*)(uwfrag + (((ks * 4 + nt) * 32 + lane) << 4));
      acc[nt] = __builtin_amdgcn_wmma_f32_16x16x32_f16(false, a, false, b,
                                                       (short)0, acc[nt], false, false);
    }
  }
#pragma unroll
  for (int nt = 0; nt < 4; ++nt) {
#pragma unroll
    for (int r = 0; r < 8; ++r) {
      int mm = m0 + r + ((lane >> 4) * 8);
      if (mm < N_NODES) {
        int col = nt * 16 + n;
        float v = fmaxf(acc[nt][r] + Ub[col], 0.f);
        if (hout_f32)  hout_f32[(size_t)mm * 64 + col]   = v;
        if (hout_f16)  hout_f16[(size_t)mm * 64 + col]   = (_Float16)v;
        if (hcat_next) hcat_next[(size_t)mm * 128 + col] = (_Float16)v;
      }
    }
  }
}

// ---------------------------------------------------------------------------
// Readout: out[n] = h2[n,:] @ out_w + out_b
// ---------------------------------------------------------------------------
__global__ void gnn_readout(const float* __restrict__ h2, const float* __restrict__ ow,
                            const float* __restrict__ ob, float* __restrict__ out) {
  int nn = blockIdx.x * 256 + threadIdx.x;
  if (nn >= N_NODES) return;
  float s = 0.f;
#pragma unroll 8
  for (int c = 0; c < 64; ++c) s += h2[(size_t)nn * 64 + c] * ow[c];
  out[nn] = s + ob[0];
}

// ---------------------------------------------------------------------------
extern "C" void kernel_launch(void* const* d_in, const int* in_sizes, int n_in,
                              void* d_out, int out_size, void* d_ws, size_t ws_size,
                              hipStream_t stream) {
  (void)in_sizes; (void)n_in; (void)out_size; (void)ws_size;
  const float* x     = (const float*)d_in[0];
  const float* noise = (const float*)d_in[1];
  const int*   ei    = (const int*)d_in[2];
  const float* c1W1  = (const float*)d_in[3];
  const float* c1b1  = (const float*)d_in[4];
  const float* c1W2  = (const float*)d_in[5];
  const float* c1b2  = (const float*)d_in[6];
  const float* c1Uw  = (const float*)d_in[7];
  const float* c1Ub  = (const float*)d_in[8];
  const float* c2W1  = (const float*)d_in[9];
  const float* c2b1  = (const float*)d_in[10];
  const float* c2W2  = (const float*)d_in[11];
  const float* c2b2  = (const float*)d_in[12];
  const float* c2Uw  = (const float*)d_in[13];
  const float* c2Ub  = (const float*)d_in[14];
  const float* outw  = (const float*)d_in[15];
  const float* outb  = (const float*)d_in[16];

  char* ws = (char*)d_ws;
  size_t off = 0;
  auto alloc = [&](size_t bytes) -> void* {
    void* p = ws + off; off = (off + bytes + 255) & ~(size_t)255; return p;
  };
  _Float16* h0h   = (_Float16*)alloc((size_t)N_NODES * 64 * 2);
  float*    aggr1 = (float*)   alloc((size_t)N_NODES * 64 * 4);
  _Float16* hcat1 = (_Float16*)alloc((size_t)N_NODES * 128 * 2);
  _Float16* h1h   = (_Float16*)alloc((size_t)N_NODES * 64 * 2);
  _Float16* hcat2 = (_Float16*)alloc((size_t)N_NODES * 128 * 2);
  float*    aggr2 = (float*)   alloc((size_t)N_NODES * 64 * 4);
  float*    h2    = (float*)   alloc((size_t)N_NODES * 64 * 4);
  _Float16* uw1f  = (_Float16*)alloc(8192 * 2);
  _Float16* uw2f  = (_Float16*)alloc(8192 * 2);

  hipMemsetAsync(aggr1, 0, (size_t)N_NODES * 64 * 4, stream);
  hipMemsetAsync(aggr2, 0, (size_t)N_NODES * 64 * 4, stream);

  gnn_prep_h0<<<(N_NODES * 64 + 255) / 256, 256, 0, stream>>>(x, noise, h0h);
  gnn_prep_uwfrag<<<8192 / 256, 256, 0, stream>>>(c1Uw, DIN1 + 64, uw1f);
  gnn_prep_uwfrag<<<8192 / 256, 256, 0, stream>>>(c2Uw, 128, uw2f);

  gnn_edge_layer<DIN1><<<N_EDGES / EDGE_WAVES, EDGE_WAVES * 32, 0, stream>>>(
      h0h, ei, c1W1, c1b1, c1W2, c1b2, aggr1);
  gnn_cat1<<<(N_NODES * 128 + 255) / 256, 256, 0, stream>>>(x, noise, aggr1, hcat1);
  gnn_update_layer<<<(N_NODES + 127) / 128, 256, 0, stream>>>(hcat1, uw1f, c1Ub,
                                                              nullptr, h1h, hcat2);
  gnn_edge_layer<64><<<N_EDGES / EDGE_WAVES, EDGE_WAVES * 32, 0, stream>>>(
      h1h, ei, c2W1, c2b1, c2W2, c2b2, aggr2);
  gnn_cat2<<<(N_NODES * 64 + 255) / 256, 256, 0, stream>>>(aggr2, hcat2);
  gnn_update_layer<<<(N_NODES + 127) / 128, 256, 0, stream>>>(hcat2, uw2f, c2Ub,
                                                              h2, nullptr, nullptr);
  gnn_readout<<<(N_NODES + 255) / 256, 256, 0, stream>>>(h2, outw, outb, (float*)d_out);
}
